// DFHLoss_20323785244686
// MI455X (gfx1250) — matrix operand order repaired and checked by reading, not compiled, not run on hardware
//
#include <hip/hip_runtime.h>
#include <hip/hip_bf16.h>
#include <cstdint>
#include <math.h>

// ---------------- CDNA5 vector types ----------------
typedef __attribute__((ext_vector_type(16))) _Float16 v16h;
typedef __attribute__((ext_vector_type(8)))  float    v8f;
typedef __attribute__((ext_vector_type(2)))  float    v2f;
typedef __attribute__((ext_vector_type(4)))  unsigned int u32x4;
typedef __attribute__((ext_vector_type(4)))  int      i32x4;
typedef __attribute__((ext_vector_type(8)))  int      i32x8;

#define BITD   64
#define NCLASS 100
#define NTRAIN 100000
#define BATCH  512
#define MU_C   1.0f
#define M_C    1.0f
#define ETA_C  0.5
#define VUL_C  1.0f
#define NTA_C  1.0f

#define NCB    391                 // ceil(100000/256) column blocks
#define NPART  (NCB * 8)           // metric partial sums

// ---------------- workspace layout (bytes) ----------------
#define OFF_U16   0                         // 512*64 f16          = 65536
#define OFF_TL    65536                     // 100000 i32          = 400000
#define OFF_LAB   465536                    // 512 i32
#define OFF_CANON 467584                    // 512 i32
#define OFF_BMAT  469632                    // 64*512 f32          = 131072
#define OFF_BS    600704                    // 64*100 f32          = 25600
#define OFF_CNT   626304                    // 100 f32
#define OFF_QUANT 626704                    // 1 double
#define OFF_MAINP 626712                    // NPART doubles       = 25024
#define OFF_CORRP 651736                    // 512 doubles
// total ~656 KB

static __device__ __forceinline__ float sgnf(float x) {
  return (x > 0.f ? 1.f : 0.f) - (x < 0.f ? 1.f : 0.f);
}
static __device__ __forceinline__ float softplusf(float x) {
  return (x > 30.f) ? x : log1pf(expf(x));
}
static __device__ __forceinline__ v8f wmma_f32_k4(v2f a, v2f b, v8f c) {
#if __has_builtin(__builtin_amdgcn_wmma_f32_16x16x4_f32)
  return __builtin_amdgcn_wmma_f32_16x16x4_f32(false, a, false, b, (short)0, c,
                                               false, false);
#else
  c[0] += a.x * b.x + a.y * b.y;  // keep compiling; layout-inexact fallback
  return c;
#endif
}

// ================= Kernel 1: prep =================
// labels from one-hot y; b = sign(mu*C@y.T + u.T); count/Bs for the collapsed
// intra gradient; canonical flags for scatter dedup; u->f16; quant loss sum.
__global__ __launch_bounds__(512) void dfh_prep(
    const float* __restrict__ u, const float* __restrict__ y,
    const int* __restrict__ ind, const float* __restrict__ V,
    _Float16* __restrict__ u16, int* __restrict__ labels,
    int* __restrict__ canon, float* __restrict__ bmat,
    float* __restrict__ Bs, float* __restrict__ cnt,
    double* __restrict__ quant) {
  __shared__ int slab[BATCH];
  __shared__ double red[16];
  int t = threadIdx.x;

  int lab = 0;
  for (int c = 0; c < NCLASS; c++)
    if (y[t * NCLASS + c] > 0.5f) lab = c;
  slab[t] = lab;
  labels[t] = lab;

  double qs = 0.0;
  for (int r = 0; r < BITD; r++) {
    float Cv = sgnf(V[r * NCLASS + lab]);
    float uval = u[t * BITD + r];
    float bb = sgnf(MU_C * Cv + uval);
    bmat[r * BATCH + t] = bb;
    float d = bb - uval;
    qs += (double)(d * d);
  }

  int my = ind[t];
  int cn = 1;
  for (int k = 0; k < t; k++)
    if (ind[k] == my) { cn = 0; break; }
  canon[t] = cn;

  for (int i = t; i < BATCH * BITD; i += 512) u16[i] = (_Float16)u[i];

  __threadfence_block();
  __syncthreads();

  if (t < NCLASS) {
    int c = 0;
    for (int k = 0; k < BATCH; k++) c += (slab[k] == t);
    cnt[t] = (float)c;
  }
  for (int e = t; e < BITD * NCLASS; e += 512) {
    int r = e / NCLASS, c = e % NCLASS;
    float s = 0.f;
    for (int k = 0; k < BATCH; k++)
      if (slab[k] == c) s += bmat[r * BATCH + k];
    Bs[e] = s;
  }

  for (int off = 16; off; off >>= 1) qs += __shfl_down(qs, off, 32);
  if ((t & 31) == 0) red[t >> 5] = qs;
  __syncthreads();
  if (t == 0) {
    double s = 0;
    for (int w = 0; w < 16; w++) s += red[w];
    *quant = s;
  }
}

// ================= Kernel 2: train labels =================
__global__ __launch_bounds__(256) void dfh_tlabel(const float* __restrict__ Y,
                                                  int* __restrict__ tl) {
  int j = blockIdx.x * 256 + threadIdx.x;
  if (j >= NTRAIN) return;
  int lab = 0;
  for (int c = 0; c < NCLASS; c++)
    if (Y[(size_t)c * NTRAIN + j] > 0.5f) lab = c;
  tl[j] = lab;
}

// ================= Kernel 3: main metric GEMM (WMMA f16 + TDM) =============
// block = 256 thr (8 waves): 64 rows x 256 cols tile.
// Each wave owns 2 col-tiles processed SEQUENTIALLY (4 row-tiles each) so at
// most 4 accumulator tiles (32 VGPRs) are live with the softplus epilogue,
// avoiding the scratch spills seen when all 8 tiles were held at once.
__global__ __launch_bounds__(256) void dfh_metric(
    const _Float16* __restrict__ u16g, const float* __restrict__ U,
    const int* __restrict__ tl, const int* __restrict__ labels,
    double* __restrict__ partials) {
  extern __shared__ char smem[];
  float*    Utile = (float*)smem;                  // [64][256]  @ 0
  _Float16* uA    = (_Float16*)(smem + 65536);     // [64][68]
  int*      labc  = (int*)(smem + 74240);          // [64]
  int*      tlc   = (int*)(smem + 74496);          // [256]
  double*   red   = (double*)(smem + 75520);       // [8]

  int tid = threadIdx.x;
  int lane = tid & 31, wave = tid >> 5;
  int lhalf = lane & 15, hi = lane >> 4;
  int colbase = blockIdx.x * 256;
  int rowbase = blockIdx.y * 64;

  for (int i = tid; i < 64 * 64; i += 256) {
    int r = i >> 6, k = i & 63;
    uA[r * 68 + k] = u16g[(rowbase + r) * 64 + k];
  }
  if (tid < 64) labc[tid] = labels[rowbase + tid];
  if (tid < 256) {
    int g = colbase + tid;
    tlc[tid] = (g < NTRAIN) ? tl[g] : -1;
  }

  // --- TDM: async DMA of the 64x256 f32 U tile into LDS (offset 0) ---
  if (tid < 32) {
    unsigned long long ga = (unsigned long long)(uintptr_t)(U + colbase);
    unsigned dim0 = (unsigned)(NTRAIN - colbase);  // OOB cols -> zero fill
    u32x4 g0 = {1u,                                // count=1
                0u,                                // lds_addr = 0
                (unsigned)(ga & 0xFFFFFFFFull),
                (((unsigned)(ga >> 32)) & 0x01FFFFFFu) | 0x80000000u};  // type=2
    i32x8 g1;
    g1[0] = (int)(2u << 16);                                   // data_size=4B
    g1[1] = (int)((dim0 & 0xFFFFu) << 16);                     // tensor_dim0 lo
    g1[2] = (int)(((dim0 >> 16) & 0xFFFFu) | (64u << 16));     // dim0 hi|dim1
    g1[3] = (int)(256u << 16);                                 // tile_dim0=256
    g1[4] = 64;                                                // tile_dim1=64
    g1[5] = (int)NTRAIN;                                       // dim0_stride
    g1[6] = 0;
    g1[7] = 0;
    i32x4 gz = {0, 0, 0, 0};
#if __clang_major__ >= 23
    i32x8 z8 = {0, 0, 0, 0, 0, 0, 0, 0};
    __builtin_amdgcn_tensor_load_to_lds(g0, g1, gz, gz, z8, 0);
#else
    __builtin_amdgcn_tensor_load_to_lds(g0, g1, gz, gz, 0);
#endif
    __builtin_amdgcn_s_wait_tensorcnt(0);
  }
  __syncthreads();

  int validCols = NTRAIN - colbase;
  if (validCols > 256) validCols = 256;

  v8f zero8 = {0, 0, 0, 0, 0, 0, 0, 0};
  double lsum = 0.0;

  for (int ct = 0; ct < 2; ct++) {
    int ctile = 2 * wave + ct;
    int col = ctile * 16 + lhalf;

    v8f acc[4];
#pragma unroll
    for (int rt = 0; rt < 4; rt++) acc[rt] = zero8;

    for (int kk = 0; kk < 2; kk++) {
      v16h bfr;
#pragma unroll
      for (int h = 0; h < 16; h++) {
        int v = h >> 1, p = h & 1;
        int k = kk * 32 + ((v < 4) ? 2 * v : 16 + 2 * (v - 4)) + 8 * hi + p;
        bfr[h] = (_Float16)Utile[k * 256 + col];
      }
#pragma unroll
      for (int rt = 0; rt < 4; rt++) {
        int row = rt * 16 + lhalf;
        v16h afr;
#pragma unroll
        for (int h = 0; h < 16; h++) {
          int v = h >> 1, p = h & 1;
          int k = kk * 32 + ((v < 4) ? 2 * v : 16 + 2 * (v - 4)) + 8 * hi + p;
          afr[h] = uA[row * 68 + k];
        }
        acc[rt] = __builtin_amdgcn_wmma_f32_16x16x32_f16(
            false, afr, false, bfr, (short)0, acc[rt], false, false);
      }
    }

    // fused softplus epilogue + reduction (never materialize ip)
    if (ctile * 16 + 16 <= validCols) {
      int tlv = tlc[ctile * 16 + lhalf];
#pragma unroll
      for (int rt = 0; rt < 4; rt++) {
        int rowl = rt * 16 + 8 * hi;
#pragma unroll
        for (int v = 0; v < 8; v++) {
          float ip = 0.5f * acc[rt][v];
          ip = fminf(fmaxf(ip, -100.f), 50.f);
          float x = (labc[rowl + v] == tlv) ? (M_C - ip) : (M_C + ip);
          lsum += (double)softplusf(x);
        }
      }
    }
  }

  for (int off = 16; off; off >>= 1) lsum += __shfl_down(lsum, off, 32);
  if (lane == 0) red[wave] = lsum;
  __syncthreads();
  if (tid == 0) {
    double s = 0;
    for (int w = 0; w < 8; w++) s += red[w];
    partials[blockIdx.y * gridDim.x + blockIdx.x] = s;
  }
}

// ================= Kernel 4: scatter correction =================
// For each first-writer k: replace column ind[k]'s contribution.
__global__ __launch_bounds__(256) void dfh_corr(
    const float* __restrict__ u, const float* __restrict__ U,
    const int* __restrict__ ind, const int* __restrict__ canon,
    const int* __restrict__ tl, const int* __restrict__ labels,
    double* __restrict__ corrp) {
  __shared__ float uk[64];
  __shared__ int info[4];
  __shared__ double red[8];
  int tid = threadIdx.x;
  int k = blockIdx.x;
  if (tid == 0) {
    int j = ind[k];
    info[0] = canon[k];
    info[1] = j;
    info[2] = labels[k];
    info[3] = tl[j];
  }
  if (tid < 64) uk[tid] = u[k * 64 + tid];
  __syncthreads();

  double ls = 0.0;
  if (info[0]) {
    int j = info[1], labk = info[2], tlj = info[3];
    for (int i = tid; i < BATCH; i += 256) {
      float dot_o = 0.f, dot_n = 0.f;
      int li = labels[i];
      for (int r = 0; r < 64; r++) {
        float ui = u[i * 64 + r];
        dot_o += ui * U[(size_t)r * NTRAIN + j];
        dot_n += ui * uk[r];
      }
      float ipo = fminf(fmaxf(0.5f * dot_o, -100.f), 50.f);
      float ipn = fminf(fmaxf(0.5f * dot_n, -100.f), 50.f);
      float fo = softplusf((li == tlj) ? (M_C - ipo) : (M_C + ipo));
      float fn = softplusf((li == labk) ? (M_C - ipn) : (M_C + ipn));
      ls += (double)(fn - fo);
    }
  }
  for (int off = 16; off; off >>= 1) ls += __shfl_down(ls, off, 32);
  if ((tid & 31) == 0) red[tid >> 5] = ls;
  __syncthreads();
  if (tid == 0) {
    double s = 0;
    for (int w = 0; w < 8; w++) s += red[w];
    corrp[k] = s;
  }
}

// ================= Kernel 5: 200-step V descent (LDS + WMMA f32 K4) ========
// intra-grad collapsed to count[c]*V - Bs; E = V^T V - TK; P = V @ E.
__global__ __launch_bounds__(512) void dfh_vstep(
    const float* __restrict__ Vin, const float* __restrict__ Bs,
    const float* __restrict__ cnt, float* __restrict__ out) {
  extern __shared__ char smem[];
  float* Vc  = (float*)smem;              // [64][112] zero-padded cols
  float* E   = (float*)(smem + 28672);    // [112][112]
  float* Bsl = (float*)(smem + 78848);    // [64][100]
  float* cl  = (float*)(smem + 104448);   // [100]

  int tid = threadIdx.x;
  int lane = tid & 31, wave = tid >> 5;
  int lhalf = lane & 15, hi = lane >> 4;

  for (int i = tid; i < 64 * 112; i += 512) {
    int r = i / 112, c = i % 112;
    Vc[i] = (c < NCLASS) ? Vin[r * NCLASS + c] : 0.f;
  }
  for (int i = tid; i < BITD * NCLASS; i += 512) Bsl[i] = Bs[i];
  if (tid < NCLASS) cl[tid] = cnt[tid];
  __syncthreads();

  const float CI = 2.0f / 32768.0f;             // 2/(64*512)
  const float CE = 4.0f * VUL_C / 10000.0f;     // 4/(100*100)
  const float CQ = 2.0f * NTA_C / 6400.0f;      // 2/(64*100)
  v8f zero8 = {0, 0, 0, 0, 0, 0, 0, 0};

  for (int it = 0; it < 200; it++) {
    // ---- E = Vc^T Vc - TK over 7x7 16x16 tiles, K=64 as 16x K=4 WMMA ----
    for (int t = wave; t < 49; t += 16) {
      int mb = (t / 7) * 16, cb = (t % 7) * 16;
      v8f a8 = zero8;
      for (int kb = 0; kb < 64; kb += 4) {
        v2f av, bv;
        av.x = Vc[(kb + 2 * hi) * 112 + mb + lhalf];
        av.y = Vc[(kb + 2 * hi + 1) * 112 + mb + lhalf];
        bv.x = Vc[(kb + 2 * hi) * 112 + cb + lhalf];
        bv.y = Vc[(kb + 2 * hi + 1) * 112 + cb + lhalf];
        a8 = wmma_f32_k4(av, bv, a8);
      }
#pragma unroll
      for (int v = 0; v < 8; v++) {
        int m = mb + v + 8 * hi, c = cb + lhalf;
        float tk = (m < NCLASS && c < NCLASS) ? ((m == c) ? 64.f : -64.f) : 0.f;
        E[m * 112 + c] = a8[v] - tk;
      }
    }
    __syncthreads();

    // ---- P = Vc @ E over 4x7 tiles, K=112 as 28x K=4 WMMA ----
    v8f pacc[2];
    int tlist[2];
    int nt = 0;
    for (int t = wave; t < 28; t += 16) {
      int rb = (t / 7) * 16, cb = (t % 7) * 16;
      v8f a8 = zero8;
      for (int kb = 0; kb < 112; kb += 4) {
        v2f av, bv;
        av.x = Vc[(rb + lhalf) * 112 + kb + 2 * hi];
        av.y = Vc[(rb + lhalf) * 112 + kb + 2 * hi + 1];
        bv.x = E[(kb + 2 * hi) * 112 + cb + lhalf];
        bv.y = E[(kb + 2 * hi + 1) * 112 + cb + lhalf];
        a8 = wmma_f32_k4(av, bv, a8);
      }
      pacc[nt] = a8;
      tlist[nt] = t;
      nt++;
    }
    __syncthreads();  // all Vc reads done before in-place update

    float alpha = 0.03f * ((it >= 149) ? 0.1f : 1.f) * ((it >= 179) ? 0.1f : 1.f);
    for (int q = 0; q < nt; q++) {
      int t = tlist[q];
      int rb = (t / 7) * 16, cb = (t % 7) * 16;
#pragma unroll
      for (int v = 0; v < 8; v++) {
        int r = rb + v + 8 * hi, c = cb + lhalf;
        if (c < NCLASS) {
          float vv = Vc[r * 112 + c];
          float G = CI * (cl[c] * vv - Bsl[r * NCLASS + c]) +
                    CE * pacc[q][v] + CQ * (vv - sgnf(vv));
          Vc[r * 112 + c] = vv - alpha * G;
        }
      }
    }
    __syncthreads();
  }

  for (int i = tid; i < BITD * NCLASS; i += 512)
    out[1 + i] = Vc[(i / NCLASS) * 112 + (i % NCLASS)];
}

// ================= Kernel 6: deterministic final reduction =================
__global__ __launch_bounds__(256) void dfh_final(
    const double* __restrict__ mainp, const double* __restrict__ corrp,
    const double* __restrict__ quant, float* __restrict__ out) {
  __shared__ double red[8];
  int tid = threadIdx.x;
  double s = 0.0;
  for (int i = tid; i < NPART; i += 256) s += mainp[i];
  for (int i = tid; i < BATCH; i += 256) s += corrp[i];
  for (int off = 16; off; off >>= 1) s += __shfl_down(s, off, 32);
  if ((tid & 31) == 0) red[tid >> 5] = s;
  __syncthreads();
  if (tid == 0) {
    double t = 0;
    for (int w = 0; w < 8; w++) t += red[w];
    double metric = t / ((double)BATCH * (double)NTRAIN);
    double q = (*quant) / ((double)BITD * (double)BATCH);
    out[0] = (float)(metric + ETA_C * q);
  }
}

// ================= launcher =================
extern "C" void kernel_launch(void* const* d_in, const int* in_sizes, int n_in,
                              void* d_out, int out_size, void* d_ws,
                              size_t ws_size, hipStream_t stream) {
  const float* u = (const float*)d_in[0];
  const float* y = (const float*)d_in[1];
  const int*   ind = (const int*)d_in[2];
  const float* U = (const float*)d_in[3];
  const float* Y = (const float*)d_in[4];
  const float* V = (const float*)d_in[5];
  float* out = (float*)d_out;

  char* ws = (char*)d_ws;
  _Float16* u16   = (_Float16*)(ws + OFF_U16);
  int*      tl    = (int*)(ws + OFF_TL);
  int*      labels= (int*)(ws + OFF_LAB);
  int*      canon = (int*)(ws + OFF_CANON);
  float*    bmat  = (float*)(ws + OFF_BMAT);
  float*    Bs    = (float*)(ws + OFF_BS);
  float*    cnt   = (float*)(ws + OFF_CNT);
  double*   quant = (double*)(ws + OFF_QUANT);
  double*   mainp = (double*)(ws + OFF_MAINP);
  double*   corrp = (double*)(ws + OFF_CORRP);

  dfh_prep<<<1, 512, 0, stream>>>(u, y, ind, V, u16, labels, canon, bmat, Bs,
                                  cnt, quant);
  dfh_tlabel<<<NCB, 256, 0, stream>>>(Y, tl);
  dfh_metric<<<dim3(NCB, 8), 256, 75584, stream>>>(u16, U, tl, labels, mainp);
  dfh_corr<<<BATCH, 256, 0, stream>>>(u, U, ind, canon, tl, labels, corrp);
  dfh_vstep<<<1, 512, 104848, stream>>>(V, Bs, cnt, out);
  dfh_final<<<1, 256, 0, stream>>>(mainp, corrp, quant, out);
}
